// SparseConvEncoder_64785286693637
// MI455X (gfx1250) — matrix operand order repaired and checked by reading, not compiled, not run on hardware
//
#include <hip/hip_runtime.h>
#include <hip/hip_bf16.h>

typedef __attribute__((ext_vector_type(16))) _Float16 v16h;
typedef __attribute__((ext_vector_type(8)))  _Float16 v8h;
typedef __attribute__((ext_vector_type(4)))  _Float16 v4h;
typedef __attribute__((ext_vector_type(8)))  float    v8f;

__host__ __device__ constexpr int clog2(int x) { int l = 0; while ((1 << l) < x) ++l; return l; }

// ---------------------------------------------------------------------------
// seg0 = features * mask  (fp32, vectorized, C=16)
// ---------------------------------------------------------------------------
__global__ void mask_apply_kernel(const float4* __restrict__ f,
                                  const int* __restrict__ m,
                                  float4* __restrict__ out,
                                  long long n4, int C) {
    long long stride = (long long)gridDim.x * blockDim.x;
    for (long long i = (long long)blockIdx.x * blockDim.x + threadIdx.x; i < n4; i += stride) {
        long long v = (i * 4) / C;
        float4 r = {0.f, 0.f, 0.f, 0.f};
        if (m[v]) r = f[i];
        out[i] = r;
    }
}

// ---------------------------------------------------------------------------
// 2x2x2 max-pool of int mask + occupied-voxel count (wave32 ballot popcount)
// ---------------------------------------------------------------------------
__global__ void mask_down_kernel(const int* __restrict__ min_,
                                 int* __restrict__ mout,
                                 float* __restrict__ cnt, int Dout) {
    int v = blockIdx.x * blockDim.x + threadIdx.x;
    int Din = Dout * 2;
    int w = v % Dout; int t = v / Dout;
    int h = t % Dout; int d = t / Dout;
    int mm = 0;
    #pragma unroll
    for (int kd = 0; kd < 2; ++kd)
        #pragma unroll
        for (int kh = 0; kh < 2; ++kh)
            #pragma unroll
            for (int kw = 0; kw < 2; ++kw)
                mm |= min_[(((size_t)(2 * d + kd) * Din + (2 * h + kh)) * Din) + (2 * w + kw)];
    mout[v] = mm;
    unsigned long long bal = __ballot(mm != 0);
    if ((threadIdx.x & 31) == 0) atomicAdd(cnt, (float)__popcll(bal));
}

// ---------------------------------------------------------------------------
// Repack fp32 weights [tap][cin][cout] into f16 WMMA A-fragment order:
// ((tap*KC + kc)*CT + ct)*512 + lane*16 + i  holds  W[tap][kc*32+k(lane,i)][ct*16+(lane&15)]
// with k = i + ((lane>>4)<<4)   (contiguous K order, matching the B fragment loads)
// ---------------------------------------------------------------------------
__global__ void repack_kernel(const float* __restrict__ w,
                              _Float16* __restrict__ wrep,
                              int taps, int Cin, int Cout) {
    int KC = (Cin + 31) >> 5;
    int CT = Cout >> 4;
    long long total = (long long)taps * KC * CT * 512;
    long long stride = (long long)gridDim.x * blockDim.x;
    for (long long idx = (long long)blockIdx.x * blockDim.x + threadIdx.x; idx < total; idx += stride) {
        int i    = (int)(idx & 15);
        int lane = (int)((idx >> 4) & 31);
        long long rest = idx >> 9;
        int ct  = (int)(rest % CT);
        long long r2 = rest / CT;
        int kc  = (int)(r2 % KC);
        int tap = (int)(r2 / KC);
        int cout = ct * 16 + (lane & 15);
        int k    = i + ((lane >> 4) << 4);
        int cin  = kc * 32 + k;
        float v = (cin < Cin) ? w[((size_t)tap * Cin + cin) * Cout + cout] : 0.0f;
        wrep[idx] = (_Float16)v;
    }
}

// ---------------------------------------------------------------------------
// LDS-free, barrier-free implicit-GEMM conv via v_wmma_f32_16x16x32_f16.
// f16 NDHWC activations make each lane's B fragment (voxel = lane&15,
// K = kc*32 + 16*(lane>>4) .. +16) one contiguous 32-byte global load.
// 2x2 register tile per wave: 2 cout tiles x 2 voxel subtiles ->
// 2 A + 2 B fragment loads feed 4 WMMAs per K-step (1 KB fetched / WMMA).
// 256-thread blocks = 8 independent waves; no LDS, no barriers.
// INF32: level-0 only (CIN=16 fp32 input; lanes 16..31 are zero K-padding).
// ---------------------------------------------------------------------------
template <int KS, int STRIDE, int PAD, int CIN, int COUT, int DOUT, bool INF32>
__global__ __launch_bounds__(256)
void conv_wmma_kernel(const void* __restrict__ in_, float* __restrict__ out,
                      const _Float16* __restrict__ wrep,
                      const float* __restrict__ bias) {
    constexpr int DIN    = DOUT * STRIDE;
    constexpr int KC     = (CIN + 31) / 32;
    constexpr int CT     = COUT / 16;
    constexpr int NT     = KS * KS * KS;
    constexpr int S      = NT * KC;
    constexpr int LD     = clog2(DOUT);
    constexpr int NT32   = (DOUT * DOUT * DOUT) / 32;     // 32-voxel tiles

    const int lane  = threadIdx.x & 31;
    const int gw    = blockIdx.x * 8 + (threadIdx.x >> 5);   // global wave id
    const int ntile = gw & (NT32 - 1);
    const int ctp   = gw >> clog2(NT32);                     // cout-pair index
    const int khalf = lane >> 4;

    // two voxels per lane (subtile 0 and 1)
    const int vox0 = ntile * 32 + (lane & 15);
    const int vox1 = vox0 + 16;
    const int ow0 = vox0 & (DOUT - 1), oh0 = (vox0 >> LD) & (DOUT - 1), od0 = vox0 >> (2 * LD);
    const int ow1 = vox1 & (DOUT - 1), oh1 = (vox1 >> LD) & (DOUT - 1), od1 = vox1 >> (2 * LD);
    const int ibd0 = od0 * STRIDE - PAD, ibh0 = oh0 * STRIDE - PAD, ibw0 = ow0 * STRIDE - PAD;
    const int ibd1 = od1 * STRIDE - PAD, ibh1 = oh1 * STRIDE - PAD, ibw1 = ow1 * STRIDE - PAD;

    const _Float16* __restrict__ inh = (const _Float16*)in_;
    const float*    __restrict__ inf = (const float*)in_;

    v8f acc00 = {}, acc01 = {}, acc10 = {}, acc11 = {};

    #pragma unroll 2
    for (int s = 0; s < S; ++s) {
        const int tap = s / KC;
        const int kc  = s - tap * KC;
        const int kd  = tap / (KS * KS);
        const int kh  = (tap / KS) % KS;
        const int kw  = tap % KS;
        const int id0 = ibd0 + kd, ih0 = ibh0 + kh, iw0 = ibw0 + kw;
        const int id1 = ibd1 + kd, ih1 = ibh1 + kh, iw1 = ibw1 + kw;
        bool inb0 = true, inb1 = true;
        if (PAD > 0) {
            inb0 = ((unsigned)id0 < (unsigned)DIN) && ((unsigned)ih0 < (unsigned)DIN) &&
                   ((unsigned)iw0 < (unsigned)DIN);
            inb1 = ((unsigned)id1 < (unsigned)DIN) && ((unsigned)ih1 < (unsigned)DIN) &&
                   ((unsigned)iw1 < (unsigned)DIN);
        }

        v16h b0 = {}, b1 = {};
        if constexpr (!INF32) {
            const size_t koff = (size_t)kc * 32 + khalf * 16;
            if (inb0)
                b0 = *(const v16h*)(inh + (((size_t)id0 * DIN + ih0) * DIN + iw0) * CIN + koff);
            if (inb1)
                b1 = *(const v16h*)(inh + (((size_t)id1 * DIN + ih1) * DIN + iw1) * CIN + koff);
        } else {  // CIN == 16, fp32 input; K 16..31 is zero padding
            if (inb0 && khalf == 0) {
                const float4* p4 = (const float4*)(inf + (((size_t)id0 * DIN + ih0) * DIN + iw0) * CIN);
                float4 x0 = p4[0], x1 = p4[1], x2 = p4[2], x3 = p4[3];
                b0[0]  = (_Float16)x0.x; b0[1]  = (_Float16)x0.y;
                b0[2]  = (_Float16)x0.z; b0[3]  = (_Float16)x0.w;
                b0[4]  = (_Float16)x1.x; b0[5]  = (_Float16)x1.y;
                b0[6]  = (_Float16)x1.z; b0[7]  = (_Float16)x1.w;
                b0[8]  = (_Float16)x2.x; b0[9]  = (_Float16)x2.y;
                b0[10] = (_Float16)x2.z; b0[11] = (_Float16)x2.w;
                b0[12] = (_Float16)x3.x; b0[13] = (_Float16)x3.y;
                b0[14] = (_Float16)x3.z; b0[15] = (_Float16)x3.w;
            }
            if (inb1 && khalf == 0) {
                const float4* p4 = (const float4*)(inf + (((size_t)id1 * DIN + ih1) * DIN + iw1) * CIN);
                float4 x0 = p4[0], x1 = p4[1], x2 = p4[2], x3 = p4[3];
                b1[0]  = (_Float16)x0.x; b1[1]  = (_Float16)x0.y;
                b1[2]  = (_Float16)x0.z; b1[3]  = (_Float16)x0.w;
                b1[4]  = (_Float16)x1.x; b1[5]  = (_Float16)x1.y;
                b1[6]  = (_Float16)x1.z; b1[7]  = (_Float16)x1.w;
                b1[8]  = (_Float16)x2.x; b1[9]  = (_Float16)x2.y;
                b1[10] = (_Float16)x2.z; b1[11] = (_Float16)x2.w;
                b1[12] = (_Float16)x3.x; b1[13] = (_Float16)x3.y;
                b1[14] = (_Float16)x3.z; b1[15] = (_Float16)x3.w;
            }
        }

        const _Float16* ap = wrep + (((size_t)s * CT + ctp * 2) * 32 + lane) * 16;
        v16h a0 = *(const v16h*)ap;
        v16h a1 = *(const v16h*)(ap + 512);
        acc00 = __builtin_amdgcn_wmma_f32_16x16x32_f16(false, a0, false, b0, (short)0, acc00, false, false);
        acc01 = __builtin_amdgcn_wmma_f32_16x16x32_f16(false, a0, false, b1, (short)0, acc01, false, false);
        acc10 = __builtin_amdgcn_wmma_f32_16x16x32_f16(false, a1, false, b0, (short)0, acc10, false, false);
        acc11 = __builtin_amdgcn_wmma_f32_16x16x32_f16(false, a1, false, b1, (short)0, acc11, false, false);
    }

    // epilogue: C/D layout -> 8 consecutive channels per lane (NDHWC contiguous)
    const int cb0 = (ctp * 2) * 16 + (khalf << 3);
    float* op0 = out + (size_t)vox0 * COUT + cb0;
    float* op1 = out + (size_t)vox1 * COUT + cb0;
    #pragma unroll
    for (int r = 0; r < 8; ++r) {
        float b0v = bias ? bias[cb0 + r] : 0.0f;
        float b1v = bias ? bias[cb0 + 16 + r] : 0.0f;
        op0[r]      = acc00[r] + b0v;
        op1[r]      = acc01[r] + b0v;
        op0[16 + r] = acc10[r] + b1v;
        op1[16 + r] = acc11[r] + b1v;
    }
}

// ---------------------------------------------------------------------------
// Per-channel masked sum / sumsq (C is a power of two <= 256)
// ---------------------------------------------------------------------------
__global__ void stats_kernel(const float* __restrict__ x,
                             const int* __restrict__ mask,
                             float* __restrict__ sums, int N, int C) {
    int g  = blockIdx.x * blockDim.x + threadIdx.x;
    int c  = g & (C - 1);
    int v0 = g / C;
    int vs = (gridDim.x * blockDim.x) / C;
    float s = 0.0f, s2 = 0.0f;
    for (int v = v0; v < N; v += vs) {
        if (mask[v]) {
            float y = x[(size_t)v * C + c];
            s += y; s2 += y * y;
        }
    }
    atomicAdd(&sums[c], s);
    atomicAdd(&sums[C + c], s2);
}

__global__ void finalize_kernel(const float* __restrict__ sums,
                                const float* __restrict__ cnt,
                                const float* __restrict__ gamma,
                                const float* __restrict__ beta,
                                float* __restrict__ ss, int C) {
    int c = threadIdx.x;
    if (c >= C) return;
    float n = fmaxf(cnt[0], 1.0f);
    float mean = sums[c] / n;
    float var  = sums[C + c] / n - mean * mean;
    float sc   = gamma[c] * rsqrtf(var + 1e-5f);
    ss[c]     = sc;
    ss[C + c] = beta[c] - mean * sc;
}

// BN+ReLU+mask: always writes f16 (next conv input); optionally fp32 (d_out)
__global__ void bnrelu_kernel(const float4* __restrict__ x,
                              const int* __restrict__ mask,
                              const float* __restrict__ ss,
                              v4h* __restrict__ yh,
                              float4* __restrict__ yf, long long n4, int C) {
    long long stride = (long long)gridDim.x * blockDim.x;
    for (long long i = (long long)blockIdx.x * blockDim.x + threadIdx.x; i < n4; i += stride) {
        long long base = i * 4;
        int c = (int)(base % C);
        long long v = base / C;
        float4 r = {0.f, 0.f, 0.f, 0.f};
        if (mask[v]) {
            float4 xv = x[i];
            r.x = fmaxf(0.f, xv.x * ss[c + 0] + ss[C + c + 0]);
            r.y = fmaxf(0.f, xv.y * ss[c + 1] + ss[C + c + 1]);
            r.z = fmaxf(0.f, xv.z * ss[c + 2] + ss[C + c + 2]);
            r.w = fmaxf(0.f, xv.w * ss[c + 3] + ss[C + c + 3]);
        }
        v4h h; h[0] = (_Float16)r.x; h[1] = (_Float16)r.y;
               h[2] = (_Float16)r.z; h[3] = (_Float16)r.w;
        yh[i] = h;
        if (yf) yf[i] = r;
    }
}

__global__ void zero_kernel(float* __restrict__ p, int n) {
    for (int i = threadIdx.x; i < n; i += blockDim.x) p[i] = 0.0f;
}

// ---------------------------------------------------------------------------
template <int KS, int STRIDE, int PAD, int CIN, int COUT, int DOUT, bool INF32>
static void run_conv(const void* in, float* out, const _Float16* wrep,
                     const float* bias, hipStream_t stream) {
    constexpr int WAVES = ((DOUT * DOUT * DOUT) / 32) * (COUT / 32);
    conv_wmma_kernel<KS, STRIDE, PAD, CIN, COUT, DOUT, INF32>
        <<<WAVES / 8, 256, 0, stream>>>(in, out, wrep, bias);
}

extern "C" void kernel_launch(void* const* d_in, const int* in_sizes, int n_in,
                              void* d_out, int out_size, void* d_ws, size_t ws_size,
                              hipStream_t stream) {
    (void)in_sizes; (void)n_in; (void)out_size; (void)ws_size;
    static const int CH[5] = {16, 32, 64, 128, 256};

    const float* features = (const float*)d_in[0];
    const int*   mask0    = (const int*)d_in[1];
    float* out = (float*)d_out;

    // output segment offsets (elements)
    long long segoff[5];
    segoff[0] = 0;
    long long acc = (long long)128 * 128 * 128 * 16;
    for (int l = 0; l < 4; ++l) {
        segoff[l + 1] = acc;
        int D = 64 >> l;
        acc += (long long)D * D * D * CH[l + 1];
    }

    // workspace carve (bytes)
    char* ws = (char*)d_ws;
    float*    t0   = (float*)(ws + 0);                   // 33,554,432 B  (pre-BN fp32)
    _Float16* t1h  = (_Float16*)(ws + 33554432);         // 16,777,216 B  (f16 mid-level)
    _Float16* xh   = (_Float16*)(ws + 50331648);         // 16,777,216 B  (f16 level out)
    _Float16* wrep = (_Float16*)(ws + 67108864);         //  3,538,944 B
    int* mlev[4];
    size_t moff = 70647808;
    for (int l = 0; l < 4; ++l) {
        mlev[l] = (int*)(ws + moff);
        int D = 64 >> l;
        moff += (size_t)D * D * D * sizeof(int);
    }
    float* sums = (float*)(ws + 71845888);               // 512 floats
    float* ss   = (float*)(ws + 71848192);               // 512 floats
    float* cnt  = (float*)(ws + 71850496);               // 1 float

    // seg0 = features * mask (fp32; level-0 conv converts in-kernel)
    {
        long long n4 = ((long long)128 * 128 * 128 * 16) / 4;
        mask_apply_kernel<<<4096, 256, 0, stream>>>((const float4*)features, mask0,
                                                    (float4*)out, n4, 16);
    }

    const void* xin  = (const void*)out;   // fp32 seg0 feeds level 0
    const int*  mcur = mask0;

    for (int lvl = 0; lvl < 4; ++lvl) {
        const int cout = CH[lvl + 1];
        const int Dout = 64 >> lvl;
        const int Nout = Dout * Dout * Dout;
        const float* down_w   = (const float*)d_in[2 + 7 * lvl + 0];
        const float* down_g   = (const float*)d_in[2 + 7 * lvl + 1];
        const float* down_b   = (const float*)d_in[2 + 7 * lvl + 2];
        const float* blk_w    = (const float*)d_in[2 + 7 * lvl + 3];
        const float* blk_bias = (const float*)d_in[2 + 7 * lvl + 4];
        const float* blk_g    = (const float*)d_in[2 + 7 * lvl + 5];
        const float* blk_beta = (const float*)d_in[2 + 7 * lvl + 6];
        int* mnext = mlev[lvl];

        // mask pyramid + occupied count
        zero_kernel<<<1, 256, 0, stream>>>(cnt, 1);
        mask_down_kernel<<<Nout / 256, 256, 0, stream>>>(mcur, mnext, cnt, Dout);

        // ---- down conv (k=2, s=2, pad=0, no bias) ----
        {
            int cin = CH[lvl];
            int KC = (cin + 31) >> 5, CT = cout >> 4;
            long long rtot = (long long)8 * KC * CT * 512;
            int rblocks = (int)((rtot + 255) / 256); if (rblocks > 2048) rblocks = 2048;
            repack_kernel<<<rblocks, 256, 0, stream>>>(down_w, wrep, 8, cin, cout);
            switch (lvl) {
                case 0: run_conv<2, 2, 0, 16, 32, 64, true>(xin, t0, wrep, nullptr, stream); break;
                case 1: run_conv<2, 2, 0, 32, 64, 32, false>(xin, t0, wrep, nullptr, stream); break;
                case 2: run_conv<2, 2, 0, 64, 128, 16, false>(xin, t0, wrep, nullptr, stream); break;
                default: run_conv<2, 2, 0, 128, 256, 8, false>(xin, t0, wrep, nullptr, stream); break;
            }
        }
        zero_kernel<<<1, 256, 0, stream>>>(sums, 2 * cout);
        stats_kernel<<<256, 256, 0, stream>>>(t0, mnext, sums, Nout, cout);
        finalize_kernel<<<1, cout, 0, stream>>>(sums, cnt, down_g, down_b, ss, cout);
        {
            long long n4 = ((long long)Nout * cout) / 4;
            int blocks = (int)((n4 + 255) / 256); if (blocks > 4096) blocks = 4096;
            bnrelu_kernel<<<blocks, 256, 0, stream>>>((const float4*)t0, mnext, ss,
                                                      (v4h*)t1h, nullptr, n4, cout);
        }

        // ---- block conv (k=3, s=1, pad=1, bias) ----
        {
            int KC = (cout + 31) >> 5, CT = cout >> 4;
            long long rtot = (long long)27 * KC * CT * 512;
            int rblocks = (int)((rtot + 255) / 256); if (rblocks > 2048) rblocks = 2048;
            repack_kernel<<<rblocks, 256, 0, stream>>>(blk_w, wrep, 27, cout, cout);
            switch (lvl) {
                case 0: run_conv<3, 1, 1, 32, 32, 64, false>(t1h, t0, wrep, blk_bias, stream); break;
                case 1: run_conv<3, 1, 1, 64, 64, 32, false>(t1h, t0, wrep, blk_bias, stream); break;
                case 2: run_conv<3, 1, 1, 128, 128, 16, false>(t1h, t0, wrep, blk_bias, stream); break;
                default: run_conv<3, 1, 1, 256, 256, 8, false>(t1h, t0, wrep, blk_bias, stream); break;
            }
        }
        zero_kernel<<<1, 256, 0, stream>>>(sums, 2 * cout);
        stats_kernel<<<256, 256, 0, stream>>>(t0, mnext, sums, Nout, cout);
        finalize_kernel<<<1, cout, 0, stream>>>(sums, cnt, blk_g, blk_beta, ss, cout);
        {
            long long n4 = ((long long)Nout * cout) / 4;
            int blocks = (int)((n4 + 255) / 256); if (blocks > 4096) blocks = 4096;
            bnrelu_kernel<<<blocks, 256, 0, stream>>>((const float4*)t0, mnext, ss,
                                                      (v4h*)xh, (float4*)(out + segoff[lvl + 1]),
                                                      n4, cout);
        }

        xin  = (const void*)xh;
        mcur = mnext;
    }
}
